// LongShortTermTransformer_78013785964700
// MI455X (gfx1250) — matrix operand order repaired and checked by reading, not compiled, not run on hardware
//
#include <hip/hip_runtime.h>
#include <stdint.h>

// ---------------------------------------------------------------------------
// Types for CDNA5 WMMA (gfx1250, wave32)
// ---------------------------------------------------------------------------
typedef __attribute__((ext_vector_type(16))) __bf16 v16bf;
typedef __attribute__((ext_vector_type(8)))  float  v8f;

union Frag16 {
  v16bf v;
  unsigned int u[8];
};

// round-to-nearest-even f32 -> bf16, two at a time packed into a dword
__device__ __forceinline__ unsigned int pack2bf(float lo, float hi) {
  unsigned int ul = __float_as_uint(lo);
  unsigned int uh = __float_as_uint(hi);
  ul += 0x7FFFu + ((ul >> 16) & 1u);
  uh += 0x7FFFu + ((uh >> 16) & 1u);
  return (ul >> 16) | (uh & 0xFFFF0000u);
}

__device__ __forceinline__ unsigned short f2bf(float f) {
  unsigned int u = __float_as_uint(f);
  u += 0x7FFFu + ((u >> 16) & 1u);
  return (unsigned short)(u >> 16);
}

__device__ __forceinline__ uint4 pack8bf(float4 x0, float4 x1) {
  return make_uint4(pack2bf(x0.x, x0.y), pack2bf(x0.z, x0.w),
                    pack2bf(x1.x, x1.y), pack2bf(x1.z, x1.w));
}

// A-matrix fragment, 16x32 bf16 (ISA 7.12.2): lane m = lane&15 (both halves),
// VGPR v holds K pair k0,k0+1 with k0 = (v<4?0:16) + (lane>>4)*8 + 2*(v&3).
__device__ __forceinline__ void load_afrag(Frag16& fa, const unsigned short* base,
                                           int rowStride, int m, int half, int kofs) {
#pragma unroll
  for (int v = 0; v < 8; ++v) {
    int k0 = kofs + ((v < 4) ? 0 : 16) + half * 8 + 2 * (v & 3);
    fa.u[v] = *(const unsigned int*)(base + m * rowStride + k0);
  }
}

// B-matrix fragment, 32x16 bf16: lane n = lane&15, VGPR v holds K pair
// k = (v<4?0:16) + (lane>>4)*8 + 2*(v&3), +1.  base holds element (n,k) at base[n*rowStride+k].
__device__ __forceinline__ void load_bfrag(Frag16& fb, const unsigned short* base,
                                           int rowStride, int n, int half, int kofs) {
#pragma unroll
  for (int v = 0; v < 8; ++v) {
    int k = kofs + ((v < 4) ? 0 : 16) + half * 8 + 2 * (v & 3);
    fb.u[v] = *(const unsigned int*)(base + n * rowStride + k);
  }
}

__device__ __forceinline__ v8f wmma_bf16(const Frag16& a, const Frag16& b, v8f c) {
  return __builtin_amdgcn_wmma_f32_16x16x32_bf16(false, a.v, false, b.v,
                                                 (short)0, c, false, false);
}

// ---------------------------------------------------------------------------
// LayerNorm over C=512, one block (256 thr) per row; optionally also emits y+add
// ---------------------------------------------------------------------------
__global__ void __launch_bounds__(256)
layernorm_k(const float* __restrict__ x, const float* __restrict__ g,
            const float* __restrict__ bta, const float* __restrict__ add,
            float* __restrict__ y0, float* __restrict__ y1, int C) {
  const int row = blockIdx.x;
  const int tid = threadIdx.x;
  const float* xr = x + (size_t)row * C;
  float a = xr[tid], c = xr[tid + 256];
  float s = a + c, s2 = a * a + c * c;
#pragma unroll
  for (int off = 16; off > 0; off >>= 1) {
    s  += __shfl_xor(s,  off, 32);
    s2 += __shfl_xor(s2, off, 32);
  }
  __shared__ float r1[8], r2[8];
  if ((tid & 31) == 0) { r1[tid >> 5] = s; r2[tid >> 5] = s2; }
  __syncthreads();
  float sum = 0.f, sq = 0.f;
#pragma unroll
  for (int i = 0; i < 8; ++i) { sum += r1[i]; sq += r2[i]; }
  const float mean = sum / (float)C;
  const float var  = sq / (float)C - mean * mean;
  const float rs   = rsqrtf(var + 1e-5f);
#pragma unroll
  for (int e = 0; e < 2; ++e) {
    int cc = tid + e * 256;
    float v = (xr[cc] - mean) * rs * g[cc] + bta[cc];
    y0[(size_t)row * C + cc] = v;
    if (y1) y1[(size_t)row * C + cc] = v + add[(size_t)row * C + cc];
  }
}

// ---------------------------------------------------------------------------
// GEMM: C[M,N] = A[M,K] * W[N,K]^T + bias (+ res).  bf16 WMMA, f32 accumulate.
// Block tile 64x64, 8 waves; wave -> 16-row tile (wave&3), 32-col strip (wave>>2).
// A is padded to gridDim.x*64 rows (workspace slots are over-allocated), so the
// staging loop is branch-free; only the epilogue (store + residual read) is
// guarded by the real row count M.  Double-buffered LDS, one barrier per K-step.
// ---------------------------------------------------------------------------
__global__ void __launch_bounds__(256)
gemm_bf16(const float* __restrict__ A, const float* __restrict__ W,
          const float* __restrict__ bias, const float* __restrict__ res,
          float* __restrict__ C, int M, int N, int K) {
  __shared__ __align__(16) unsigned short As[2][64 * 32];
  __shared__ __align__(16) unsigned short Bs[2][64 * 32];
  const int tid  = threadIdx.x;
  const int lane = tid & 31, wave = tid >> 5;
  const int half = lane >> 4, l16 = lane & 15;
  const int bm = blockIdx.x * 64, bn = blockIdx.y * 64;
  const int wm = (wave & 3) * 16;
  const int wn = (wave >> 2) * 32;
  const int ar = tid >> 2, ac = (tid & 3) * 8;     // staging coords (64 x 32)
  const float* aptr = A + (size_t)(bm + ar) * K + ac;
  const float* wptr = W + (size_t)(bn + ar) * K + ac;
  const int sofs = ar * 32 + ac;
  v8f acc0 = {}; v8f acc1 = {};

  // prologue: stage tile 0
  {
    float4 a0 = ((const float4*)aptr)[0];
    float4 a1 = ((const float4*)aptr)[1];
    float4 w0 = ((const float4*)wptr)[0];
    float4 w1 = ((const float4*)wptr)[1];
    *(uint4*)&As[0][sofs] = pack8bf(a0, a1);
    *(uint4*)&Bs[0][sofs] = pack8bf(w0, w1);
  }
  __syncthreads();
  int buf = 0;
  for (int kt = 32; kt < K; kt += 32) {
    // prefetch next tile from global while the matrix pipe works
    float4 a0 = ((const float4*)(aptr + kt))[0];
    float4 a1 = ((const float4*)(aptr + kt))[1];
    float4 w0 = ((const float4*)(wptr + kt))[0];
    float4 w1 = ((const float4*)(wptr + kt))[1];
    Frag16 fa, fb0, fb1;
    load_afrag(fa,  As[buf] + wm * 32, 32, l16, half, 0);
    load_bfrag(fb0, Bs[buf] + wn * 32, 32, l16, half, 0);
    load_bfrag(fb1, Bs[buf] + (wn + 16) * 32, 32, l16, half, 0);
    acc0 = wmma_bf16(fa, fb0, acc0);
    acc1 = wmma_bf16(fa, fb1, acc1);
    buf ^= 1;
    *(uint4*)&As[buf][sofs] = pack8bf(a0, a1);
    *(uint4*)&Bs[buf][sofs] = pack8bf(w0, w1);
    __syncthreads();
  }
  {   // last tile
    Frag16 fa, fb0, fb1;
    load_afrag(fa,  As[buf] + wm * 32, 32, l16, half, 0);
    load_bfrag(fb0, Bs[buf] + wn * 32, 32, l16, half, 0);
    load_bfrag(fb1, Bs[buf] + (wn + 16) * 32, 32, l16, half, 0);
    acc0 = wmma_bf16(fa, fb0, acc0);
    acc1 = wmma_bf16(fa, fb1, acc1);
  }
#pragma unroll
  for (int r = 0; r < 8; ++r) {
    int row = bm + wm + r + 8 * half;
    if (row < M) {
      int c0 = bn + wn + l16;
      float v0 = acc0[r] + bias[c0];
      float v1 = acc1[r] + bias[c0 + 16];
      if (res) {
        v0 += res[(size_t)row * N + c0];
        v1 += res[(size_t)row * N + c0 + 16];
      }
      C[(size_t)row * N + c0]      = v0;
      C[(size_t)row * N + c0 + 16] = v1;
    }
  }
}

// ---------------------------------------------------------------------------
// Flash attention with WMMA.  Grid: (ceil(L/128), B*NH).  Block 256 = 8 waves,
// each wave owns 16 query rows.  Q pre-scaled by 1/8 (hd=64).
// Layouts: Q/K/V/O are [L,B,C] with head slice at h*64.
// ---------------------------------------------------------------------------
__global__ void __launch_bounds__(256)
flash_attn(const float* __restrict__ Qg, const float* __restrict__ Kg,
           const float* __restrict__ Vg, float* __restrict__ Og,
           int Lq, int Lk, int Bt, int Cch) {
  __shared__ __align__(16) unsigned short Qs[128 * 64];
  __shared__ __align__(16) unsigned short Ks[32 * 64];
  __shared__ __align__(16) unsigned short Vs[64 * 32];   // transposed: [d][key]
  __shared__ __align__(16) unsigned short Ps[8 * 16 * 32];
  const int tid  = threadIdx.x;
  const int lane = tid & 31, wave = tid >> 5;
  const int half = lane >> 4, l16 = lane & 15;
  const int h = blockIdx.y & 7, b = blockIdx.y >> 3;
  const int qb = blockIdx.x * 128;
  const int cofs = h * 64;

  {   // stage scaled Q: 128 x 64, wide loads/stores, one guard per thread
    int r = tid >> 1, c0 = (tid & 1) * 32;
    int q = qb + r;
    uint4* dst = (uint4*)&Qs[r * 64 + c0];
    if (q < Lq) {
      const float4* src = (const float4*)(Qg + ((size_t)q * Bt + b) * Cch + cofs + c0);
#pragma unroll
      for (int i = 0; i < 4; ++i) {
        float4 x0 = src[2 * i], x1 = src[2 * i + 1];
        dst[i] = make_uint4(pack2bf(x0.x * 0.125f, x0.y * 0.125f),
                            pack2bf(x0.z * 0.125f, x0.w * 0.125f),
                            pack2bf(x1.x * 0.125f, x1.y * 0.125f),
                            pack2bf(x1.z * 0.125f, x1.w * 0.125f));
      }
    } else {
      uint4 z = make_uint4(0u, 0u, 0u, 0u);
#pragma unroll
      for (int i = 0; i < 4; ++i) dst[i] = z;
    }
  }
  __syncthreads();
  Frag16 aq0, aq1;
  load_afrag(aq0, Qs + wave * 16 * 64, 64, l16, half, 0);
  load_afrag(aq1, Qs + wave * 16 * 64, 64, l16, half, 32);

  float m8[8], l8[8];
  v8f zero = {};
  v8f oacc[4];
#pragma unroll
  for (int r = 0; r < 8; ++r) { m8[r] = -1e30f; l8[r] = 0.f; }
#pragma unroll
  for (int j = 0; j < 4; ++j) oacc[j] = zero;

  for (int kb = 0; kb < Lk; kb += 32) {
    {   // stage K row-major: 32 keys x 64 d
      int r = tid >> 3, c0 = (tid & 7) * 8;
      int kk = kb + r;
      uint4* dst = (uint4*)&Ks[r * 64 + c0];
      if (kk < Lk) {
        const float4* src = (const float4*)(Kg + ((size_t)kk * Bt + b) * Cch + cofs + c0);
        float4 x0 = src[0], x1 = src[1];
        *dst = pack8bf(x0, x1);
      } else {
        *dst = make_uint4(0u, 0u, 0u, 0u);
      }
    }
    {   // stage V transposed [d][key]: thread owns key-pair x 4 d values,
        // packs (key,key+1) bf16 pairs = B-fragment's contiguous K pair.
      int kp = tid & 15, d0 = (tid >> 4) * 4;
      int k0 = kb + 2 * kp, k1 = k0 + 1;
      float4 va = make_float4(0.f, 0.f, 0.f, 0.f);
      float4 vb = va;
      if (k0 < Lk) va = *(const float4*)(Vg + ((size_t)k0 * Bt + b) * Cch + cofs + d0);
      if (k1 < Lk) vb = *(const float4*)(Vg + ((size_t)k1 * Bt + b) * Cch + cofs + d0);
      unsigned int* vdst = (unsigned int*)Vs;
      vdst[(d0 + 0) * 16 + kp] = pack2bf(va.x, vb.x);
      vdst[(d0 + 1) * 16 + kp] = pack2bf(va.y, vb.y);
      vdst[(d0 + 2) * 16 + kp] = pack2bf(va.z, vb.z);
      vdst[(d0 + 3) * 16 + kp] = pack2bf(va.w, vb.w);
    }
    __syncthreads();

    // S = Qs . K^T  (two 16-key subtiles, K(dim)=64 -> 2 wmma each)
    v8f s0 = zero, s1 = zero;
    Frag16 fb;
    load_bfrag(fb, Ks, 64, l16, half, 0);       s0 = wmma_bf16(aq0, fb, s0);
    load_bfrag(fb, Ks, 64, l16, half, 32);      s0 = wmma_bf16(aq1, fb, s0);
    load_bfrag(fb, Ks, 64, 16 + l16, half, 0);  s1 = wmma_bf16(aq0, fb, s1);
    load_bfrag(fb, Ks, 64, 16 + l16, half, 32); s1 = wmma_bf16(aq1, fb, s1);

    const bool ok0 = (kb + l16) < Lk;
    const bool ok1 = (kb + 16 + l16) < Lk;
#pragma unroll
    for (int r = 0; r < 8; ++r) {
      if (!ok0) s0[r] = -1e30f;
      if (!ok1) s1[r] = -1e30f;
    }
    // online softmax per row (rows r+8*half, 16-lane reductions stay in-half)
#pragma unroll
    for (int r = 0; r < 8; ++r) {
      float t = fmaxf(s0[r], s1[r]);
      t = fmaxf(t, __shfl_xor(t, 1, 32));
      t = fmaxf(t, __shfl_xor(t, 2, 32));
      t = fmaxf(t, __shfl_xor(t, 4, 32));
      t = fmaxf(t, __shfl_xor(t, 8, 32));
      float nm   = fmaxf(m8[r], t);
      float corr = __expf(m8[r] - nm);
      m8[r] = nm;
      float p0 = __expf(s0[r] - nm);
      float p1 = __expf(s1[r] - nm);
      s0[r] = p0; s1[r] = p1;
      float rs = p0 + p1;
      rs += __shfl_xor(rs, 1, 32);
      rs += __shfl_xor(rs, 2, 32);
      rs += __shfl_xor(rs, 4, 32);
      rs += __shfl_xor(rs, 8, 32);
      l8[r] = l8[r] * corr + rs;
      oacc[0][r] *= corr; oacc[1][r] *= corr;
      oacc[2][r] *= corr; oacc[3][r] *= corr;
    }
    // P (C-layout) -> LDS -> A-fragment
    unsigned short* pw = Ps + wave * 16 * 32;
#pragma unroll
    for (int r = 0; r < 8; ++r) {
      pw[(r + 8 * half) * 32 + l16]      = f2bf(s0[r]);
      pw[(r + 8 * half) * 32 + 16 + l16] = f2bf(s1[r]);
    }
    __syncthreads();
    Frag16 pa;
    load_afrag(pa, pw, 32, l16, half, 0);
#pragma unroll
    for (int j = 0; j < 4; ++j) {
      Frag16 bv;
      load_bfrag(bv, Vs + (j * 16) * 32, 32, l16, half, 0);
      oacc[j] = wmma_bf16(pa, bv, oacc[j]);
    }
    __syncthreads();
  }
#pragma unroll
  for (int r = 0; r < 8; ++r) {
    int q = qb + wave * 16 + r + 8 * half;
    if (q < Lq) {
      float inv = 1.0f / l8[r];
      size_t base = ((size_t)q * Bt + b) * Cch + cofs;
#pragma unroll
      for (int j = 0; j < 4; ++j)
        Og[base + j * 16 + l16] = oacc[j][r] * inv;
    }
  }
}

// ---------------------------------------------------------------------------
// Local 15x15 dilated window attention with relative-position logits.
// Grid: (B*NH, ceil(1296/256)).  logit = q . (k/8 + rel_w[p]) + rel_b[p].
// ---------------------------------------------------------------------------
__global__ void __launch_bounds__(256)
local_attn(const float* __restrict__ Qp, const float* __restrict__ Kp,
           const float* __restrict__ Vp, const float* __restrict__ relw_g,
           const float* __restrict__ relb_g, float* __restrict__ out,
           int Bt, int Cch) {
  __shared__ float relw[225 * 64];
  __shared__ float relb[225];
  const int h = blockIdx.x & 7, b = blockIdx.x >> 3;
  for (int i = threadIdx.x; i < 225 * 64; i += 256)
    relw[i] = relw_g[(size_t)h * 225 * 64 + i];
  for (int i = threadIdx.x; i < 225; i += 256)
    relb[i] = relb_g[h * 225 + i];
  __syncthreads();

  int pix = blockIdx.y * 256 + threadIdx.x;
  if (pix >= 1296) return;
  const int pi = pix / 36, pj = pix % 36;
  float q[64];
  const size_t qbase = ((size_t)pix * Bt + b) * Cch + h * 64;
#pragma unroll
  for (int d = 0; d < 64; ++d) q[d] = Qp[qbase + d];
  float m = -1e30f, l = 0.f;
  float acc[64];
#pragma unroll
  for (int d = 0; d < 64; ++d) acc[d] = 0.f;

  for (int p = 0; p < 225; ++p) {
    int oi = p / 15, oj = p % 15;
    int ki = pi + oi - 7, kj = pj + oj - 7;
    if (ki < 0 || ki >= 36 || kj < 0 || kj >= 36) continue;
    size_t kb = ((size_t)(ki * 36 + kj) * Bt + b) * Cch + h * 64;
    const float4* kp4 = (const float4*)(Kp + kb);
    const float4* rw4 = (const float4*)&relw[p * 64];
    float logit = relb[p];
#pragma unroll
    for (int dq = 0; dq < 16; ++dq) {
      float4 kv = kp4[dq];
      float4 rv = rw4[dq];
      logit += q[4 * dq + 0] * fmaf(kv.x, 0.125f, rv.x);
      logit += q[4 * dq + 1] * fmaf(kv.y, 0.125f, rv.y);
      logit += q[4 * dq + 2] * fmaf(kv.z, 0.125f, rv.z);
      logit += q[4 * dq + 3] * fmaf(kv.w, 0.125f, rv.w);
    }
    float nm = fmaxf(m, logit);
    float f  = __expf(m - nm);
    float w  = __expf(logit - nm);
    m = nm;
    l = l * f + w;
    const float4* vp4 = (const float4*)(Vp + kb);
#pragma unroll
    for (int dq = 0; dq < 16; ++dq) {
      float4 vv = vp4[dq];
      acc[4 * dq + 0] = acc[4 * dq + 0] * f + w * vv.x;
      acc[4 * dq + 1] = acc[4 * dq + 1] * f + w * vv.y;
      acc[4 * dq + 2] = acc[4 * dq + 2] * f + w * vv.z;
      acc[4 * dq + 3] = acc[4 * dq + 3] * f + w * vv.w;
    }
  }
  float inv = 1.0f / l;
  size_t ob = ((size_t)pix * Bt + b) * Cch + h * 64;
#pragma unroll
  for (int d = 0; d < 64; ++d) out[ob + d] = acc[d] * inv;
}

// ---------------------------------------------------------------------------
// GroupNorm(32) + exact GELU.  Grid: B*32 blocks; writes [B,F,H,W] layout.
// ---------------------------------------------------------------------------
__global__ void __launch_bounds__(256)
groupnorm_gelu(const float* __restrict__ hbuf, const float* __restrict__ gg,
               const float* __restrict__ gb, float* __restrict__ xg,
               int Bt, int Fc, int HW) {
  const int b = blockIdx.x >> 5, g = blockIdx.x & 31;
  const int cpg = Fc / 32;
  const int ch0 = g * cpg;
  const int n = cpg * HW;
  float s = 0.f, s2 = 0.f;
  for (int i = threadIdx.x; i < n; i += 256) {
    int c = i / HW, l = i % HW;
    float v = hbuf[((size_t)l * Bt + b) * Fc + ch0 + c];
    s += v; s2 += v * v;
  }
#pragma unroll
  for (int off = 16; off > 0; off >>= 1) {
    s  += __shfl_xor(s,  off, 32);
    s2 += __shfl_xor(s2, off, 32);
  }
  __shared__ float r1[8], r2[8];
  if ((threadIdx.x & 31) == 0) { r1[threadIdx.x >> 5] = s; r2[threadIdx.x >> 5] = s2; }
  __syncthreads();
  float sum = 0.f, sq = 0.f;
#pragma unroll
  for (int i = 0; i < 8; ++i) { sum += r1[i]; sq += r2[i]; }
  const float mean = sum / (float)n;
  const float var  = sq / (float)n - mean * mean;
  const float rs   = rsqrtf(var + 1e-5f);
  for (int i = threadIdx.x; i < n; i += 256) {
    int c = i / HW, l = i % HW;
    int f = ch0 + c;
    float v = hbuf[((size_t)l * Bt + b) * Fc + f];
    v = (v - mean) * rs * gg[f] + gb[f];
    v = 0.5f * v * (1.0f + erff(v * 0.70710678118654752f));
    xg[((size_t)b * Fc + f) * HW + l] = v;
  }
}

// ---------------------------------------------------------------------------
// 5x5 depthwise conv, pad 2; reads [B,F,H,W], writes [L,B,F].
// ---------------------------------------------------------------------------
__global__ void __launch_bounds__(256)
dwconv5x5(const float* __restrict__ xg, const float* __restrict__ w,
          float* __restrict__ y, int Bt, int Fc, int H, int W) {
  int idx = blockIdx.x * 256 + threadIdx.x;
  int total = Bt * Fc * H * W;
  if (idx >= total) return;
  int pix = idx % (H * W);
  int bf  = idx / (H * W);
  int f = bf % Fc, b = bf / Fc;
  int i = pix / W, j = pix % W;
  const float* xp = xg + (size_t)bf * H * W;
  const float* wp = w + (size_t)f * 25;
  float acc = 0.f;
#pragma unroll
  for (int di = 0; di < 5; ++di) {
    int ii = i + di - 2;
    if (ii < 0 || ii >= H) continue;
#pragma unroll
    for (int dj = 0; dj < 5; ++dj) {
      int jj = j + dj - 2;
      if (jj < 0 || jj >= W) continue;
      acc += xp[ii * W + jj] * wp[di * 5 + dj];
    }
  }
  y[((size_t)pix * Bt + b) * Fc + f] = acc;
}

// ---------------------------------------------------------------------------
// Host launch
// ---------------------------------------------------------------------------
extern "C" void kernel_launch(void* const* d_in, const int* in_sizes, int n_in,
                              void* d_out, int out_size, void* d_ws, size_t ws_size,
                              hipStream_t stream) {
  (void)in_sizes; (void)n_in; (void)out_size; (void)ws_size;
  const float* tgt      = (const float*)d_in[0];
  const float* id_emb   = (const float*)d_in[1];
  const float* self_pos = (const float*)d_in[2];
  const float* n1g = (const float*)d_in[3],  *n1b = (const float*)d_in[4];
  const float* n2g = (const float*)d_in[5],  *n2b = (const float*)d_in[6];
  const float* n3g = (const float*)d_in[7],  *n3b = (const float*)d_in[8];
  const float* sa_qw = (const float*)d_in[9],  *sa_qb = (const float*)d_in[10];
  const float* sa_kw = (const float*)d_in[11], *sa_kb = (const float*)d_in[12];
  const float* sa_vw = (const float*)d_in[13], *sa_vb = (const float*)d_in[14];
  const float* sa_pw = (const float*)d_in[15], *sa_pb = (const float*)d_in[16];
  const float* lq_w  = (const float*)d_in[17], *lq_b  = (const float*)d_in[18];
  const float* lv_w  = (const float*)d_in[19], *lv_b  = (const float*)d_in[20];
  const float* lt_pw = (const float*)d_in[21], *lt_pb = (const float*)d_in[22];
  const float* st_rel_w = (const float*)d_in[23], *st_rel_b = (const float*)d_in[24];
  const float* st_pw = (const float*)d_in[25], *st_pb = (const float*)d_in[26];
  const float* f1w = (const float*)d_in[27], *f1b = (const float*)d_in[28];
  const float* gn_g = (const float*)d_in[29], *gn_b = (const float*)d_in[30];
  const float* dw_w = (const float*)d_in[31];
  const float* f2w = (const float*)d_in[32], *f2b = (const float*)d_in[33];

  const int L = 1296, B = 2, C = 512, F = 2048, M = L * B;  // M = 2592
  const int MP = ((M + 63) / 64) * 64;                      // 2624 padded rows
  float* ws = (float*)d_ws;
  const size_t LBCp = (size_t)MP * C;   // padded slot strides
  const size_t LBFp = (size_t)MP * F;
  float* S0 = ws + 0 * LBCp;
  float* S1 = ws + 1 * LBCp;
  float* S2 = ws + 2 * LBCp;
  float* S3 = ws + 3 * LBCp;
  float* S4 = ws + 4 * LBCp;
  float* S5 = ws + 5 * LBCp;
  float* S6 = ws + 6 * LBCp;
  float* B0 = ws + 7 * LBCp;
  float* B1 = B0 + LBFp;

  auto gemm = [&](const float* A, const float* W, const float* bias,
                  const float* res, float* Cc, int Nn, int Kk) {
    dim3 grid(MP / 64, Nn / 64);   // compute over padded rows, store guarded by M
    gemm_bf16<<<grid, 256, 0, stream>>>(A, W, bias, res, Cc, M, Nn, Kk);
  };
  const dim3 fgrid((L + 127) / 128, B * 8);

  // ---- self attention ----
  layernorm_k<<<M, 256, 0, stream>>>(tgt, n1g, n1b, self_pos, S0, S1, C); // S0=_t, S1=qkx
  gemm(S1, sa_qw, sa_qb, nullptr, S2, C, C);             // Q
  gemm(S1, sa_kw, sa_kb, nullptr, S3, C, C);             // K
  gemm(S0, sa_vw, sa_vb, nullptr, S4, C, C);             // V
  flash_attn<<<fgrid, 256, 0, stream>>>(S2, S3, S4, S5, L, L, B, C);
  gemm(S5, sa_pw, sa_pb, tgt, S6, C, C);                 // S6 = t1 = tgt + proj

  // ---- long/short term attention ----
  layernorm_k<<<M, 256, 0, stream>>>(S6, n2g, n2b, id_emb, S0, S1, C); // S0=_t2, S1=_t2+id
  gemm(S0, lq_w, lq_b, nullptr, S2, C, C);               // curr_Q (= curr_K)
  gemm(S1, lv_w, lv_b, nullptr, S3, C, C);               // global_V
  flash_attn<<<fgrid, 256, 0, stream>>>(S2, S2, S3, S4, L, L, B, C);
  local_attn<<<dim3(B * 8, (L + 255) / 256), 256, 0, stream>>>(S2, S2, S3,
                                                               st_rel_w, st_rel_b,
                                                               S5, B, C);
  gemm(S4, lt_pw, lt_pb, S6, S1, C, C);                  // S1 = t1 + lt proj
  gemm(S5, st_pw, st_pb, S1, S6, C, C);                  // S6 = pre-FFN residual

  // ---- GN-gated FFN ----
  layernorm_k<<<M, 256, 0, stream>>>(S6, n3g, n3b, nullptr, S0, nullptr, C);
  gemm(S0, f1w, f1b, nullptr, B0, F, C);                 // h [M,2048]
  groupnorm_gelu<<<B * 32, 256, 0, stream>>>(B0, gn_g, gn_b, B1, B, F, L);
  {
    int total = B * F * L;
    dwconv5x5<<<(total + 255) / 256, 256, 0, stream>>>(B1, dw_w, B0, B, F, 36, 36);
  }
  gemm(B0, f2w, f2b, S6, (float*)d_out, C, F);           // out = S6 + ffn
}
